// HermitePolynomials_13597866459390
// MI455X (gfx1250) — compile-verified, matrix operand order checked by reading, not run on hardware
//
#include <hip/hip_runtime.h>
#include <stdint.h>

// ---------------------------------------------------------------------------
// Hermite-feature GEMM for MI455X (gfx1250, wave32, WMMA)
//   y[n,o] = sum_{i,d} H_d(tanh(x[n,i])) * C[i,o,d]
// Flatten k = i*8+d  ->  GEMM  [32768 x 2048] * [2048 x 256]
// f16 WMMA (V_WMMA_F32_16X16X32_F16), f32 accumulate.
//   - f16 chosen over bf16: same XDL shape/rate, 1-instr packed convert
//     (v_cvt_pk_rtz_f16_f32), and more mantissa for tanh-bounded features.
// B tile staged via GLOBAL_LOAD_ASYNC_TO_LDS_B128 (ASYNCcnt path);
// A tile (tanh + Hermite recurrence) generated in-register and ds_store'd.
// ---------------------------------------------------------------------------

typedef __attribute__((ext_vector_type(16))) _Float16 v16h;
typedef __attribute__((ext_vector_type(8)))  float    v8f;
typedef __attribute__((ext_vector_type(4)))  int      v4i;

#define IN_DIM   256
#define OUT_DIM  256
#define DEG      8                 // degree+1
#define K_TOTAL  (IN_DIM * DEG)    // 2048
#define N_ROWS   (16 * 2048)       // 32768

#define BLK_N    128               // rows per workgroup
#define BLK_O    128               // output cols per workgroup
#define KC       64                // K chunk (8 input channels * 8 degrees)
#define LDST     72                // padded LDS row stride (elements; 144B -> bank-spread, 16B aligned)

// ---------------- feature helpers (CDNA5-native where available) -----------

static __device__ __forceinline__ float fast_tanh(float x) {
#if __has_builtin(__builtin_amdgcn_tanhf)
    return __builtin_amdgcn_tanhf(x);          // v_tanh_f32
#else
    // branchless: tanh(x) = 1 - 2/(e^{2x}+1); exp2 saturation gives correct +-1 tails
    float e = __builtin_amdgcn_exp2f(x * 2.8853900817779268f);   // 2*log2(e)
    return 1.0f - 2.0f * __builtin_amdgcn_rcpf(e + 1.0f);
#endif
}

static __device__ __forceinline__ uint32_t pack_f16(float lo, float hi) {
#if __has_builtin(__builtin_amdgcn_cvt_pkrtz)
    typedef __attribute__((ext_vector_type(2))) __fp16 v2fp16;   // builtin's return type
    v2fp16 r = __builtin_amdgcn_cvt_pkrtz(lo, hi);   // v_cvt_pk_rtz_f16_f32 (1 instr)
    return __builtin_bit_cast(uint32_t, r);
#else
    uint16_t a = __builtin_bit_cast(uint16_t, (_Float16)lo);
    uint16_t b = __builtin_bit_cast(uint16_t, (_Float16)hi);
    return (uint32_t)a | ((uint32_t)b << 16);
#endif
}

#if __has_builtin(__builtin_amdgcn_global_load_async_to_lds_b128)
#define HAVE_ASYNC_LDS 1
// builtin expects (as1 v4i*, as3 v4i*, imm offset, imm cpol) per hipcc diagnostic
typedef __attribute__((address_space(1))) v4i as1_v4i;
typedef __attribute__((address_space(3))) v4i as3_v4i;
#else
#define HAVE_ASYNC_LDS 0
#endif

// ---------------------------------------------------------------------------
// Kernel 1: repack coeffs C[i][o][d] (f32) -> Bg[o][i*8+d] (f16, row-major K)
// ---------------------------------------------------------------------------
__global__ __launch_bounds__(256) void prep_B(const float* __restrict__ C,
                                              uint16_t* __restrict__ Bg) {
    int idx = blockIdx.x * 256 + threadIdx.x;   // 0 .. 65535
    int i = idx >> 8;                           // input channel
    int o = idx & 255;                          // output channel
    const float* src = C + ((size_t)i * OUT_DIM + o) * DEG;
    uint32_t p0 = pack_f16(src[0], src[1]);
    uint32_t p1 = pack_f16(src[2], src[3]);
    uint32_t p2 = pack_f16(src[4], src[5]);
    uint32_t p3 = pack_f16(src[6], src[7]);
    *(uint4*)(Bg + (size_t)o * K_TOTAL + (size_t)i * DEG) = make_uint4(p0, p1, p2, p3);
}

// ---------------------------------------------------------------------------
// Kernel 2: fused tanh+Hermite feature generation + f16 WMMA GEMM.
// 256 threads = 8 waves; wave grid 4(row) x 2(col); each wave: 2x4 16x16 tiles.
// ---------------------------------------------------------------------------
__global__ __launch_bounds__(256) void hermite_gemm(const float* __restrict__ X,
                                                    const uint16_t* __restrict__ Bg,
                                                    float* __restrict__ Y) {
    __shared__ uint16_t Alds[BLK_N * LDST];   // features, [row][k-in-chunk]
    __shared__ uint16_t Blds[BLK_O * LDST];   // coeffs,   [ocol][k-in-chunk]

    const int tid     = threadIdx.x;
    const int lane    = tid & 31;
    const int wave    = tid >> 5;        // 0..7
    const int waveRow = wave >> 1;       // 0..3 -> rows waveRow*32
    const int waveCol = wave & 1;        // 0..1 -> cols waveCol*64

    const int rowBlock = blockIdx.x * BLK_N;
    const int colBlock = blockIdx.y * BLK_O;

    const int l15   = lane & 15;
    const int lhalf = lane >> 4;         // 0 | 1
    const int klo   = lhalf * 8;         // 16-bit operand layout: lanes>=16 hold K 8-15 / 24-31

    v8f acc[2][4];
#pragma unroll
    for (int rt = 0; rt < 2; ++rt)
#pragma unroll
        for (int ct = 0; ct < 4; ++ct)
            acc[rt][ct] = v8f{0.f, 0.f, 0.f, 0.f, 0.f, 0.f, 0.f, 0.f};

    for (int kc = 0; kc < K_TOTAL / KC; ++kc) {   // 32 chunks of K=64
        __syncthreads();   // protect LDS reuse from previous chunk

        // ---- stage B: Bg[colBlock..+127][kc*64..+63] -> LDS
#if HAVE_ASYNC_LDS
#pragma unroll
        for (int r = 0; r < 4; ++r) {
            int e  = tid + r * 256;      // 0..1023 groups of 8 f16
            int o  = e >> 3;             // 0..127
            int kg = e & 7;
            as1_v4i* gp = (as1_v4i*)(Bg + (size_t)(colBlock + o) * K_TOTAL + kc * KC + kg * 8);
            as3_v4i* lp = (as3_v4i*)&Blds[o * LDST + kg * 8];
            __builtin_amdgcn_global_load_async_to_lds_b128(gp, lp, 0, 0);
        }
#else
#pragma unroll
        for (int r = 0; r < 4; ++r) {
            int e  = tid + r * 256;
            int o  = e >> 3;
            int kg = e & 7;
            uint4 v = *(const uint4*)(Bg + (size_t)(colBlock + o) * K_TOTAL + kc * KC + kg * 8);
            *(uint4*)&Blds[o * LDST + kg * 8] = v;
        }
#endif

        // ---- stage A: x[128 rows][8 channels] -> tanh -> H0..H7 -> f16 LDS
        // 1024 x-elements, 4 per thread, one 16B LDS store per element
#pragma unroll
        for (int r = 0; r < 4; ++r) {
            int e   = tid + r * 256;     // 0..1023
            int row = e >> 3;
            int ch  = e & 7;
            float xv = X[(size_t)(rowBlock + row) * IN_DIM + kc * 8 + ch];
            float t  = fast_tanh(xv);
            float t2 = 2.0f * t;
            float h[DEG];
            h[0] = 1.0f;
            h[1] = t2;
#pragma unroll
            for (int d = 2; d < DEG; ++d)
                h[d] = t2 * h[d - 1] - 2.0f * (float)(d - 1) * h[d - 2];
            uint32_t p0 = pack_f16(h[0], h[1]);
            uint32_t p1 = pack_f16(h[2], h[3]);
            uint32_t p2 = pack_f16(h[4], h[5]);
            uint32_t p3 = pack_f16(h[6], h[7]);
            *(uint4*)&Alds[row * LDST + ch * 8] = make_uint4(p0, p1, p2, p3);
        }

#if HAVE_ASYNC_LDS
#if __has_builtin(__builtin_amdgcn_s_wait_asynccnt)
        __builtin_amdgcn_s_wait_asynccnt(0);
#else
        asm volatile("s_wait_asynccnt 0" ::: "memory");
#endif
#endif
        __syncthreads();

        // ---- compute: two k-steps of 32 per chunk
#pragma unroll
        for (int ks = 0; ks < 2; ++ks) {
            const int kbase = ks * 32 + klo;

            union Frag { v16h v; uint4 u[2]; };
            Frag afr[2], bfr[4];
            // A fragment: lane = row (M); two 8-element K runs at klo and klo+16
#pragma unroll
            for (int rt = 0; rt < 2; ++rt) {
                const uint16_t* p = &Alds[(waveRow * 32 + rt * 16 + l15) * LDST + kbase];
                afr[rt].u[0] = *(const uint4*)p;
                afr[rt].u[1] = *(const uint4*)(p + 16);
            }
            // B fragment: lane = output column (N); same K striping
#pragma unroll
            for (int ct = 0; ct < 4; ++ct) {
                const uint16_t* p = &Blds[(waveCol * 64 + ct * 16 + l15) * LDST + kbase];
                bfr[ct].u[0] = *(const uint4*)p;
                bfr[ct].u[1] = *(const uint4*)(p + 16);
            }
#pragma unroll
            for (int rt = 0; rt < 2; ++rt)
#pragma unroll
                for (int ct = 0; ct < 4; ++ct)
                    acc[rt][ct] = __builtin_amdgcn_wmma_f32_16x16x32_f16(
                        /*neg_a=*/false, afr[rt].v,
                        /*neg_b=*/false, bfr[ct].v,
                        /*c_mod=*/(short)0, acc[rt][ct],
                        /*reuse_a=*/false, /*reuse_b=*/false);
        }
    }

    // ---- writeback: C/D layout: element v of acc -> row (v + 8*lhalf), col = lane&15
#pragma unroll
    for (int rt = 0; rt < 2; ++rt) {
        int rbase = rowBlock + waveRow * 32 + rt * 16 + lhalf * 8;
#pragma unroll
        for (int ct = 0; ct < 4; ++ct) {
            int col = colBlock + waveCol * 64 + ct * 16 + l15;
#pragma unroll
            for (int v = 0; v < 8; ++v)
                Y[(size_t)(rbase + v) * OUT_DIM + col] = acc[rt][ct][v];
        }
    }
}

// ---------------------------------------------------------------------------
extern "C" void kernel_launch(void* const* d_in, const int* in_sizes, int n_in,
                              void* d_out, int out_size, void* d_ws, size_t ws_size,
                              hipStream_t stream) {
    const float* X = (const float*)d_in[0];          // [16,2048,256] f32
    const float* C = (const float*)d_in[1];          // [256,256,8]   f32
    float*       Y = (float*)d_out;                  // [16,2048,256] f32
    uint16_t*    Bg = (uint16_t*)d_ws;               // 2048*256 f16 = 1 MB scratch

    // repack coeffs -> f16 B matrix [O=256][K=2048]
    prep_B<<<dim3((IN_DIM * OUT_DIM) / 256), 256, 0, stream>>>(C, Bg);

    // fused feature-gen + GEMM
    hermite_gemm<<<dim3(N_ROWS / BLK_N, OUT_DIM / BLK_O), 256, 0, stream>>>(X, Bg, Y);
}
